// QuantizerBase_39797166964972
// MI455X (gfx1250) — matrix operand order repaired and checked by reading, not compiled, not run on hardware
//
#include <hip/hip_runtime.h>

typedef __attribute__((ext_vector_type(16))) __bf16 v16bf;
typedef __attribute__((ext_vector_type(8)))  float  v8f;

#define N_Q   32768
#define N_K   8192
#define DIM   512
#define KT    16      // 512 / 32 k-tiles
#define RT    2048    // 32768 / 16 q row-tiles
#define CT    512     // 8192 / 16 codebook col-tiles

// ---------------------------------------------------------------------------
// Prepass: codebook row norms  ||k_m||^2
// ---------------------------------------------------------------------------
__global__ void knorm_kernel(const float* __restrict__ k, float* __restrict__ knorm) {
    __shared__ float red[256];
    int row = blockIdx.x, t = threadIdx.x;
    float s = 0.f;
    for (int c = t; c < DIM; c += 256) { float v = k[(size_t)row * DIM + c]; s += v * v; }
    red[t] = s; __syncthreads();
    for (int off = 128; off; off >>= 1) { if (t < off) red[t] += red[t + off]; __syncthreads(); }
    if (t == 0) knorm[row] = red[0];
}

// ---------------------------------------------------------------------------
// Prepass: f32 -> bf16, swizzled into WMMA A-fragment layout (16x32, 16-bit)
// qsw chunk id = (rt*16 + kt)*32 + lane, 16 bf16 per chunk (32B contiguous)
// ---------------------------------------------------------------------------
__global__ void swz_q_kernel(const float* __restrict__ q, __bf16* __restrict__ qsw) {
    int id   = blockIdx.x * blockDim.x + threadIdx.x;   // RT*KT*32 threads
    int lane = id & 31;
    int kt   = (id >> 5) & 15;
    int rt   = id >> 9;
    int row  = rt * 16 + (lane & 15);
    int half = lane >> 4;
    const float* src = q + (size_t)row * DIM + kt * 32;
    v16bf val;
#pragma unroll
    for (int i = 0; i < 16; ++i) {
        int v = i >> 1, p = i & 1;
        int kk = 16 * half + 2 * v + p;
        val[i] = (__bf16)src[kk];
    }
    ((v16bf*)qsw)[id] = val;
}

// B packing (32x16, 16-bit), contiguous k order as in A: lane<16: N=lane, K=0..15; lane>=16: K=16..31
__global__ void swz_k_kernel(const float* __restrict__ k, __bf16* __restrict__ ksw) {
    int id   = blockIdx.x * blockDim.x + threadIdx.x;   // CT*KT*32 threads
    int lane = id & 31;
    int kt   = (id >> 5) & 15;
    int ct   = id >> 9;
    int row  = ct * 16 + (lane & 15);
    const float* src = k + (size_t)row * DIM + kt * 32 + 16 * (lane >> 4);
    v16bf val;
#pragma unroll
    for (int i = 0; i < 16; ++i) val[i] = (__bf16)src[i];
    ((v16bf*)ksw)[id] = val;
}

// ---------------------------------------------------------------------------
// Fused GEMM + argmin with double-buffered async global->LDS B staging.
// ---------------------------------------------------------------------------
__device__ __forceinline__ void async_fill_tile(unsigned lds_addr, unsigned long long gaddr) {
    // each lane copies 64B: 4 x b128, same IOFFSET applies to LDS and memory sides
    asm volatile("global_load_async_to_lds_b128 %0, %1, off"
                 :: "v"(lds_addr), "v"(gaddr) : "memory");
    asm volatile("global_load_async_to_lds_b128 %0, %1, off offset:4096"
                 :: "v"(lds_addr), "v"(gaddr) : "memory");
    asm volatile("global_load_async_to_lds_b128 %0, %1, off offset:8192"
                 :: "v"(lds_addr), "v"(gaddr) : "memory");
    asm volatile("global_load_async_to_lds_b128 %0, %1, off offset:12288"
                 :: "v"(lds_addr), "v"(gaddr) : "memory");
}

__device__ __forceinline__ void wait_async0() {
#if __has_builtin(__builtin_amdgcn_s_wait_asynccnt)
    __builtin_amdgcn_s_wait_asynccnt(0);
#else
    asm volatile("s_wait_asynccnt 0x0" ::: "memory");
#endif
}

__global__ __launch_bounds__(256, 1)
void argmin_kernel(const __bf16* __restrict__ qsw,
                   const __bf16* __restrict__ ksw,
                   const float*  __restrict__ knorm,
                   int* __restrict__ idx_out) {
    __shared__ alignas(32) __bf16 smem[2 * KT * 32 * 16];   // 2 x 16 KB double buffer
    const int tid  = threadIdx.x;
    const int lane = tid & 31;
    const int wave = tid >> 5;
    const int rt   = blockIdx.x * 8 + wave;

    // Load all 16 A fragments for this row-tile (32B contiguous per lane each)
    const v16bf* aptr = (const v16bf*)qsw + (size_t)rt * (KT * 32) + lane;
    v16bf a[KT];
#pragma unroll
    for (int kt = 0; kt < KT; ++kt) a[kt] = aptr[kt * 32];

    float minv[8]; int mini[8];
#pragma unroll
    for (int j = 0; j < 8; ++j) { minv[j] = 3.4e38f; mini[j] = 0; }

    const unsigned lds_base = (unsigned)(unsigned long long)(&smem[0]);
    const unsigned my_lds   = lds_base + (unsigned)tid * 16u;
    unsigned long long ga   = (unsigned long long)ksw + (unsigned long long)tid * 16ull;

    async_fill_tile(my_lds, ga);            // preload tile 0 into buffer 0

    for (int ct = 0; ct < CT; ++ct) {
        wait_async0();                      // tile ct resident in buffer ct&1
        __syncthreads();                    // all waves see it; prev buffer free

        if (ct + 1 < CT) {                  // overlap: fetch tile ct+1
            ga += 16384ull;
            async_fill_tile(lds_base + (unsigned)((ct + 1) & 1) * 16384u +
                            (unsigned)tid * 16u, ga);
        }

        const float kn = knorm[ct * 16 + (lane & 15)];
        const v16bf* bfr = (const v16bf*)(smem + (ct & 1) * (KT * 32 * 16)) + lane;

        v8f acc0 = {0.f, 0.f, 0.f, 0.f, 0.f, 0.f, 0.f, 0.f};
        v8f acc1 = {0.f, 0.f, 0.f, 0.f, 0.f, 0.f, 0.f, 0.f};
#pragma unroll
        for (int kt = 0; kt < KT; kt += 2) {    // two chains break the WMMA RAW chain
            v16bf b0 = bfr[kt * 32];
            v16bf b1 = bfr[(kt + 1) * 32];
            acc0 = __builtin_amdgcn_wmma_f32_16x16x32_bf16(
                       false, a[kt], false, b0, (short)0, acc0, false, false);
            acc1 = __builtin_amdgcn_wmma_f32_16x16x32_bf16(
                       false, a[kt + 1], false, b1, (short)0, acc1, false, false);
        }

        const int nidx = ct * 16 + (lane & 15);
#pragma unroll
        for (int j = 0; j < 8; ++j) {
            float score = kn - 2.0f * (acc0[j] + acc1[j]);
            if (score < minv[j]) { minv[j] = score; mini[j] = nidx; }
        }
        // no trailing barrier needed: next iteration's wait+barrier fences reuse
    }

    // C layout: lane L holds N=L%16, M = 8*(L/16)+j  -> reduce over 16 N-lanes
#pragma unroll
    for (int j = 0; j < 8; ++j) {
        float v = minv[j]; int ii = mini[j];
#pragma unroll
        for (int off = 8; off; off >>= 1) {
            float ov = __shfl_xor(v, off, 32);
            int   oi = __shfl_xor(ii, off, 32);
            if (ov < v || (ov == v && oi < ii)) { v = ov; ii = oi; }
        }
        if ((lane & 15) == 0) {
            int row = rt * 16 + (lane >> 4) * 8 + j;
            idx_out[row] = ii;
        }
    }
}

// ---------------------------------------------------------------------------
// Epilogue kernels
// ---------------------------------------------------------------------------
__global__ void fill_zero_kernel(float4* __restrict__ onehot4, unsigned int* __restrict__ counts,
                                 long n4) {
    long id     = (long)blockIdx.x * blockDim.x + threadIdx.x;
    long stride = (long)gridDim.x * blockDim.x;
    float4 z = {0.f, 0.f, 0.f, 0.f};
    for (long i = id; i < n4; i += stride) onehot4[i] = z;
    if (id < N_K) counts[id] = 0u;
}

__global__ void scatter_kernel(const int* __restrict__ idx, float* __restrict__ out_idx,
                               float* __restrict__ onehot, unsigned int* __restrict__ counts) {
    int r = blockIdx.x * blockDim.x + threadIdx.x;
    if (r < N_Q) {
        int m = idx[r];
        out_idx[r] = (float)m;
        onehot[(size_t)r * N_K + m] = 1.0f;
        atomicAdd(&counts[m], 1u);
    }
}

__global__ void gather_z_kernel(const int* __restrict__ idx, const float* __restrict__ k,
                                float* __restrict__ z) {
    long id = (long)blockIdx.x * blockDim.x + threadIdx.x;
    if (id < (long)N_Q * DIM) {
        long row = id >> 9;
        int  c   = (int)(id & (DIM - 1));
        z[id] = k[(size_t)idx[row] * DIM + c];
    }
}

__global__ void perplexity_kernel(const unsigned int* __restrict__ counts,
                                  float* __restrict__ out) {
    __shared__ float red[256];
    int t = threadIdx.x;
    float s = 0.f;
    for (int m = t; m < N_K; m += 256) {
        float p = (float)counts[m] * (1.0f / (float)N_Q);
        s += p * __logf(p + 1e-10f);
    }
    red[t] = s; __syncthreads();
    for (int off = 128; off; off >>= 1) { if (t < off) red[t] += red[t + off]; __syncthreads(); }
    if (t == 0) out[0] = __expf(-red[0]);
}

// ---------------------------------------------------------------------------
extern "C" void kernel_launch(void* const* d_in, const int* in_sizes, int n_in,
                              void* d_out, int out_size, void* d_ws, size_t ws_size,
                              hipStream_t stream) {
    const float* q = (const float*)d_in[0];   // [32768, 512]
    const float* k = (const float*)d_in[1];   // [8192, 512]

    // Output layout (flat floats, reference return order)
    float* out        = (float*)d_out;
    float* out_z      = out;                               // 32768*512
    float* out_idx    = out + (size_t)N_Q * DIM;           // 32768
    float* out_onehot = out_idx + N_Q;                     // 32768*8192
    float* out_ppl    = out_onehot + (size_t)N_Q * N_K;    // 1

    // Workspace layout
    char* ws = (char*)d_ws;
    __bf16*       qsw    = (__bf16*)ws;                                    // 32 MB
    __bf16*       ksw    = (__bf16*)(ws + ((size_t)32 << 20));             //  8 MB
    float*        knorm  = (float*)(ws + ((size_t)40 << 20));              // 32 KB
    int*          idxbuf = (int*)(ws + ((size_t)40 << 20) + (32 << 10));   // 128 KB
    unsigned int* counts = (unsigned int*)(ws + ((size_t)40 << 20) + (160 << 10)); // 32 KB

    knorm_kernel<<<N_K, 256, 0, stream>>>(k, knorm);
    swz_q_kernel<<<(RT * KT * 32) / 256, 256, 0, stream>>>(q, qsw);
    swz_k_kernel<<<(CT * KT * 32) / 256, 256, 0, stream>>>(k, ksw);

    argmin_kernel<<<RT / 8, 256, 0, stream>>>(qsw, ksw, knorm, idxbuf);

    long n4 = ((long)N_Q * N_K) / 4;
    fill_zero_kernel<<<32768, 256, 0, stream>>>((float4*)out_onehot, counts, n4);
    scatter_kernel<<<(N_Q + 255) / 256, 256, 0, stream>>>(idxbuf, out_idx, out_onehot, counts);
    gather_z_kernel<<<(int)(((long)N_Q * DIM) / 256), 256, 0, stream>>>(idxbuf, k, out_z);
    perplexity_kernel<<<1, 256, 0, stream>>>(counts, out_ppl);

    (void)in_sizes; (void)n_in; (void)out_size; (void)ws_size;
}
